// GCNNet_48326972014947
// MI455X (gfx1250) — compile-verified, matrix operand order checked by reading, not run on hardware
//
#include <hip/hip_runtime.h>

typedef __attribute__((ext_vector_type(16))) __bf16 v16bf;
typedef __attribute__((ext_vector_type(8)))  __bf16 v8bf;
typedef __attribute__((ext_vector_type(8)))  float  v8f;

// ---------------------------------------------------------------- utilities
__global__ void k_zero(uint32_t* p, size_t n) {
    size_t i = (size_t)blockIdx.x * blockDim.x + threadIdx.x;
    if (i < n) p[i] = 0u;
}

__global__ void k_fillf(float* p, float v, size_t n) {
    size_t i = (size_t)blockIdx.x * blockDim.x + threadIdx.x;
    if (i < n) p[i] = v;
}

__global__ void k_cvt(const float* __restrict__ s, __bf16* __restrict__ d, size_t n, int relu) {
    size_t i = (size_t)blockIdx.x * blockDim.x + threadIdx.x;
    if (i < n) {
        float v = s[i];
        if (relu) v = fmaxf(v, 0.0f);
        d[i] = (__bf16)v;
    }
}

// W[K x Nn] fp32 (row-major) -> Wt[Np x Kp] bf16 transposed, zero-padded
__global__ void k_prep_w(const float* __restrict__ W, __bf16* __restrict__ Wt,
                         int K, int Nn, int Kp, int Np) {
    size_t t = (size_t)blockIdx.x * blockDim.x + threadIdx.x;
    size_t tot = (size_t)Kp * Np;
    if (t >= tot) return;
    int n = (int)(t / Kp);
    int k = (int)(t % Kp);
    float v = (k < K && n < Nn) ? W[(size_t)k * Nn + n] : 0.0f;
    Wt[t] = (__bf16)v;
}

// x[R x F] fp32 -> xb[R x Fp] bf16, zero-padded
__global__ void k_prep_x(const float* __restrict__ x, __bf16* __restrict__ xb,
                         int R, int F, int Fp) {
    size_t t = (size_t)blockIdx.x * blockDim.x + threadIdx.x;
    size_t tot = (size_t)R * Fp;
    if (t >= tot) return;
    int i = (int)(t / Fp);
    int f = (int)(t % Fp);
    xb[t] = (__bf16)((f < F) ? x[(size_t)i * F + f] : 0.0f);
}

// ---------------------------------------------------------------- GCN pieces
__global__ void k_deg(const int* __restrict__ ei, float* deg, int E) {
    int e = blockIdx.x * blockDim.x + threadIdx.x;
    if (e < E) atomicAdd(deg + ei[E + e], 1.0f);   // ei row1 = dst
}

__global__ void k_rsqrt(float* p, int n) {
    int i = blockIdx.x * blockDim.x + threadIdx.x;
    if (i < n) p[i] = rsqrtf(p[i]);
}

// agg[dst] += h[src] * dinv[src]*dinv[dst]
__global__ void k_scatter(const float* __restrict__ h, const int* __restrict__ ei,
                          const float* __restrict__ dinv, float* __restrict__ agg,
                          int E, int Fp) {
    int e = blockIdx.x;
    int src = ei[e], dst = ei[E + e];
    float w = dinv[src] * dinv[dst];
    const float* hs = h + (size_t)src * Fp;
    float* ad = agg + (size_t)dst * Fp;
    for (int f = threadIdx.x; f < Fp; f += blockDim.x)
        atomicAdd(ad + f, hs[f] * w);
}

// xout = relu(agg + h*dinv^2 + b) as bf16
__global__ void k_combine(const float* __restrict__ h, const float* __restrict__ agg,
                          const float* __restrict__ dinv, const float* __restrict__ bias,
                          __bf16* __restrict__ xout, int Fp, int Fact) {
    int i = blockIdx.x;
    float di = dinv[i];
    float d2 = di * di;
    size_t base = (size_t)i * Fp;
    for (int f = threadIdx.x; f < Fp; f += blockDim.x) {
        float v = agg[base + f] + h[base + f] * d2 + ((f < Fact) ? bias[f] : 0.0f);
        xout[base + f] = (__bf16)fmaxf(v, 0.0f);
    }
}

// segment max over nodes (values >= 0 after relu) via int-bit atomicMax
__global__ void k_pool(const __bf16* __restrict__ x, float* __restrict__ gmax,
                       int Fp, int nodesPerSeg) {
    int i = blockIdx.x;
    int seg = i / nodesPerSeg;
    size_t xo = (size_t)i * Fp;
    size_t go = (size_t)seg * Fp;
    for (int f = threadIdx.x; f < Fp; f += blockDim.x) {
        float v = (float)x[xo + f];           // v >= 0
        atomicMax((int*)(gmax + go + f), __float_as_int(v));
    }
}

// row L2-normalize cell, emit bf16 zero-padded
__global__ void k_cellnorm(const float* __restrict__ cell, __bf16* __restrict__ out,
                           int F, int Fp) {
    __shared__ float red[256];
    int row = blockIdx.x;
    const float* cr = cell + (size_t)row * F;
    float s = 0.0f;
    for (int f = threadIdx.x; f < F; f += 256) { float v = cr[f]; s += v * v; }
    red[threadIdx.x] = s;
    __syncthreads();
    for (int st = 128; st > 0; st >>= 1) {
        if (threadIdx.x < st) red[threadIdx.x] += red[threadIdx.x + st];
        __syncthreads();
    }
    float rn = 1.0f / fmaxf(sqrtf(red[0]), 1e-12f);
    __bf16* orow = out + (size_t)row * Fp;
    for (int f = threadIdx.x; f < Fp; f += 256)
        orow[f] = (__bf16)((f < F) ? cr[f] * rn : 0.0f);
}

__global__ void k_concat(const float* __restrict__ g1, const float* __restrict__ g2,
                         const float* __restrict__ cv, __bf16* __restrict__ xc, int B) {
    int t = blockIdx.x * blockDim.x + threadIdx.x;
    if (t >= B * 384) return;
    int row = t / 384, c = t % 384;
    float v;
    if (c < 128)      v = g1[row * 128 + c];
    else if (c < 256) v = g2[row * 128 + (c - 128)];
    else              v = cv[row * 128 + (c - 256)];
    xc[t] = (__bf16)v;
}

// ---------------------------------------------------------------- WMMA GEMM
// C[M x Np] (fp32, ldc) = A[M x Kp] (bf16) @ Bt[Np x Kp]^T (bf16) + bias.
// Requirements: Np % 64 == 0 (gridDim.x = Np/64), M % 128 == 0 (gridDim.y = M/128),
// Kp % 32 == 0. Block = 128 threads (4 waves). Each wave computes 32 rows x 64 cols:
// 2 A-fragments x 4 B-fragments -> 8 accumulators, 12 b128 loads : 8 WMMA per K step.
// Split-K via blockIdx.z (kchunk K-range per z); accum=1 -> atomicAdd into zeroed C,
// bias applied only by z==0, relu must be 0 (applied later in k_cvt).
// A fragment per CDNA5 16-bit A layout: lane<16 holds K {0..7,16..23}+kb,
// lane>=16 holds K {8..15,24..31}+kb, two 16B contiguous loads each.
__global__ __launch_bounds__(128)
void k_gemm(const __bf16* __restrict__ A, const __bf16* __restrict__ Bt,
            const float* __restrict__ bias, float* __restrict__ C,
            int Kp, int kchunk, int ldc, int nact, int nbias, int relu, int accum) {
    const int lane = threadIdx.x & 31;
    const int wave = threadIdx.x >> 5;
    const int l    = lane & 15;
    const int hi   = lane >> 4;
    const int m0   = blockIdx.y * 128 + wave * 32;
    const int nb   = blockIdx.x * 64;

    int k0 = blockIdx.z * kchunk;
    int k1 = k0 + kchunk;
    if (k1 > Kp) k1 = Kp;

    v8f acc[2][4] = {};

    union F { v16bf v; v8bf h[2]; };

    const __bf16* arow0 = A + (size_t)(m0 + l) * Kp + hi * 8;
    const __bf16* arow1 = arow0 + (size_t)16 * Kp;
    const __bf16* brow  = Bt + (size_t)(nb + l) * Kp + hi * 8;

    for (int kb = k0; kb < k1; kb += 32) {
        F a0, a1, bm[4];
        a0.h[0] = *(const v8bf*)(arow0 + kb);
        a0.h[1] = *(const v8bf*)(arow0 + kb + 16);
        a1.h[0] = *(const v8bf*)(arow1 + kb);
        a1.h[1] = *(const v8bf*)(arow1 + kb + 16);
        __builtin_prefetch(arow0 + kb + 64, 0, 1);   // global_prefetch_b8
#pragma unroll
        for (int j = 0; j < 4; ++j) {
            const __bf16* bp = brow + (size_t)(j * 16) * Kp + kb;
            bm[j].h[0] = *(const v8bf*)(bp);
            bm[j].h[1] = *(const v8bf*)(bp + 16);
        }
#pragma unroll
        for (int j = 0; j < 4; ++j) {
            acc[0][j] = __builtin_amdgcn_wmma_f32_16x16x32_bf16(
                false, a0.v, false, bm[j].v, (short)0, acc[0][j], false, false);
            acc[1][j] = __builtin_amdgcn_wmma_f32_16x16x32_bf16(
                false, a1.v, false, bm[j].v, (short)0, acc[1][j], false, false);
        }
    }

    const int z0 = (blockIdx.z == 0);
#pragma unroll
    for (int i = 0; i < 2; ++i) {
#pragma unroll
        for (int j = 0; j < 4; ++j) {
            int col = nb + j * 16 + l;
            if (col < nact) {
                float bv = (bias != nullptr && col < nbias && z0) ? bias[col] : 0.0f;
#pragma unroll
                for (int r = 0; r < 8; ++r) {
                    int row = m0 + i * 16 + hi * 8 + r;   // C/D layout: VGPR r -> M=r (+8 hi lanes)
                    float v = acc[i][j][r] + bv;
                    if (accum) {
                        atomicAdd(C + (size_t)row * ldc + col, v);
                    } else {
                        if (relu) v = fmaxf(v, 0.0f);
                        C[(size_t)row * ldc + col] = v;
                    }
                }
            }
        }
    }
}

// ---------------------------------------------------------------- host side
extern "C" void kernel_launch(void* const* d_in, const int* in_sizes, int n_in,
                              void* d_out, int out_size, void* d_ws, size_t ws_size,
                              hipStream_t stream) {
    (void)in_sizes; (void)n_in; (void)out_size; (void)ws_size;

    const float* x1  = (const float*)d_in[0];
    const int*   ei1 = (const int*)d_in[1];
    const float* x2  = (const float*)d_in[3];
    const int*   ei2 = (const int*)d_in[4];
    const float* cell= (const float*)d_in[6];
    const float* Wc1 = (const float*)d_in[7];  const float* bc1 = (const float*)d_in[8];
    const float* Wc2 = (const float*)d_in[9];  const float* bc2 = (const float*)d_in[10];
    const float* Wc3 = (const float*)d_in[11]; const float* bc3 = (const float*)d_in[12];
    const float* Wg1 = (const float*)d_in[13]; const float* bg1 = (const float*)d_in[14];
    const float* Wg2 = (const float*)d_in[15]; const float* bg2 = (const float*)d_in[16];
    const float* Wr1 = (const float*)d_in[17]; const float* br1 = (const float*)d_in[18];
    const float* Wr2 = (const float*)d_in[19]; const float* br2 = (const float*)d_in[20];
    const float* Wr3 = (const float*)d_in[21]; const float* br3 = (const float*)d_in[22];
    const float* Wf1 = (const float*)d_in[23]; const float* bf1_ = (const float*)d_in[24];
    const float* Wf2 = (const float*)d_in[25]; const float* bf2_ = (const float*)d_in[26];
    const float* Wo  = (const float*)d_in[27]; const float* bo  = (const float*)d_in[28];
    float* out = (float*)d_out;

    constexpr int N = 40960, E = 131072, B = 1024;
    constexpr int FXT = 37261, FXTp = 37280;

    // ---- workspace bump allocator (small, persistent buffers) ----
    char* base = (char*)d_ws; size_t off = 0;
    auto alloc = [&](size_t bytes) -> void* {
        off = (off + 255) & ~(size_t)255;
        void* p = base + off; off += bytes; return p;
    };
    __bf16* Wc1t = (__bf16*)alloc((size_t)128 * 96  * 2);
    __bf16* Wc2t = (__bf16*)alloc((size_t)192 * 128 * 2);
    __bf16* Wc3t = (__bf16*)alloc((size_t)320 * 192 * 2);
    __bf16* Wg1t = (__bf16*)alloc((size_t)192 * 320 * 2);
    __bf16* Wg2t = (__bf16*)alloc((size_t)128 * 192 * 2);
    __bf16* Wr2t = (__bf16*)alloc((size_t)256 * 512 * 2);
    __bf16* Wr3t = (__bf16*)alloc((size_t)128 * 256 * 2);
    __bf16* Wf1t = (__bf16*)alloc((size_t)512 * 384 * 2);
    __bf16* Wf2t = (__bf16*)alloc((size_t)128 * 512 * 2);
    __bf16* Wot  = (__bf16*)alloc((size_t)64  * 128 * 2);
    float*  dinv  = (float*)alloc((size_t)N * 4);
    float*  gmax  = (float*)alloc((size_t)B * 320 * 4);
    __bf16* gmaxb = (__bf16*)alloc((size_t)B * 320 * 2);
    float*  t1    = (float*)alloc((size_t)B * 192 * 4);
    __bf16* t1b   = (__bf16*)alloc((size_t)B * 192 * 2);
    float*  g1    = (float*)alloc((size_t)B * 128 * 4);
    float*  g2    = (float*)alloc((size_t)B * 128 * 4);
    float*  cv1   = (float*)alloc((size_t)B * 512 * 4);
    __bf16* cv1b  = (__bf16*)alloc((size_t)B * 512 * 2);
    float*  cv2   = (float*)alloc((size_t)B * 256 * 4);
    __bf16* cv2b  = (__bf16*)alloc((size_t)B * 256 * 2);
    float*  cvv   = (float*)alloc((size_t)B * 128 * 4);
    __bf16* xcb   = (__bf16*)alloc((size_t)B * 384 * 2);
    float*  f1    = (float*)alloc((size_t)B * 512 * 4);
    __bf16* f1b   = (__bf16*)alloc((size_t)B * 512 * 2);
    float*  f2    = (float*)alloc((size_t)B * 128 * 4);
    __bf16* f2b   = (__bf16*)alloc((size_t)B * 128 * 2);

    // ---- big union region: cell branch first, GCN region reuses it ----
    size_t big0 = (off + 255) & ~(size_t)255;
    // region A (cell phase)
    __bf16* cellN = (__bf16*)(base + big0);
    __bf16* Wr1t  = (__bf16*)(base + big0 + (size_t)B * FXTp * 2);
    // region B (GCN phase) — aliases region A, used only after cell branch done
    __bf16* xa  = (__bf16*)(base + big0);
    __bf16* xb  = (__bf16*)(base + big0 + (size_t)N * 320 * 2);
    float*  h   = (float*)(base + big0 + (size_t)N * 320 * 2 * 2);
    float*  agg = (float*)(base + big0 + (size_t)N * 320 * 2 * 2 + (size_t)N * 320 * 4);

    auto zeroW = [&](void* p, size_t words) {
        k_zero<<<(unsigned)((words + 255) / 256), 256, 0, stream>>>((uint32_t*)p, words);
    };
    auto cvt = [&](const float* s, __bf16* d, size_t n, int relu) {
        k_cvt<<<(unsigned)((n + 255) / 256), 256, 0, stream>>>(s, d, n, relu);
    };
    auto prepw = [&](const float* W, __bf16* Wt, int K, int Nn, int Kp, int Np) {
        size_t n = (size_t)Kp * Np;
        k_prep_w<<<(unsigned)((n + 255) / 256), 256, 0, stream>>>(W, Wt, K, Nn, Kp, Np);
    };
    auto gemm = [&](const __bf16* A, const __bf16* Bt, const float* bias, float* C,
                    int M, int Kp, int Np, int ldc, int nact, int nbias, int relu,
                    int chunks, int accum) {
        int steps = Kp / 32;
        int kchunk = ((steps + chunks - 1) / chunks) * 32;
        dim3 g((unsigned)(Np / 64), (unsigned)(M / 128), (unsigned)chunks);
        k_gemm<<<g, 128, 0, stream>>>(A, Bt, bias, C, Kp, kchunk, ldc, nact, nbias, relu, accum);
    };

    // ---- small weight preps (all N padded to x64, K padded to prev-layer pad) ----
    prepw(Wc1, Wc1t, 78,  78,  96,  128);
    prepw(Wc2, Wc2t, 78,  156, 128, 192);
    prepw(Wc3, Wc3t, 156, 312, 192, 320);
    prepw(Wg1, Wg1t, 312, 156, 320, 192);
    prepw(Wg2, Wg2t, 156, 128, 192, 128);
    prepw(Wr2, Wr2t, 512, 256, 512, 256);
    prepw(Wr3, Wr3t, 256, 128, 256, 128);
    prepw(Wf1, Wf1t, 384, 512, 384, 512);
    prepw(Wf2, Wf2t, 512, 128, 512, 128);
    prepw(Wo,  Wot,  128, 2,   128, 64);

    // ======== cell-line branch (region A) ========
    k_cellnorm<<<B, 256, 0, stream>>>(cell, cellN, FXT, FXTp);
    prepw(Wr1, Wr1t, FXT, 512, FXTp, 512);
    // dominant GEMM: 1024 x 37280 x 512, split-K over 8 chunks for parallelism
    zeroW(cv1, (size_t)B * 512);
    gemm(cellN, Wr1t, br1, cv1, B, FXTp, 512, 512, 512, 512, /*relu=*/0, /*chunks=*/8, /*accum=*/1);
    cvt(cv1, cv1b, (size_t)B * 512, /*relu=*/1);
    gemm(cv1b, Wr2t, br2, cv2, B, 512, 256, 256, 256, 256, 1, 1, 0);
    cvt(cv2, cv2b, (size_t)B * 256, 0);
    gemm(cv2b, Wr3t, br3, cvv, B, 256, 128, 128, 128, 128, 0, 1, 0);

    // ======== drug branches (region B; both use drug1 weights, like the torch code) ========
    auto branch = [&](const float* x, const int* ei, float* gout) {
        size_t nx = (size_t)N * 96;
        k_prep_x<<<(unsigned)((nx + 255) / 256), 256, 0, stream>>>(x, xa, N, 78, 96);
        k_fillf<<<(N + 255) / 256, 256, 0, stream>>>(dinv, 1.0f, (size_t)N);
        k_deg<<<(E + 255) / 256, 256, 0, stream>>>(ei, dinv, E);
        k_rsqrt<<<(N + 255) / 256, 256, 0, stream>>>(dinv, N);
        // layer1: xa(K=96) -> h(N=128)
        gemm(xa, Wc1t, nullptr, h, N, 96, 128, 128, 128, 0, 0, 1, 0);
        zeroW(agg, (size_t)N * 128);
        k_scatter<<<E, 128, 0, stream>>>(h, ei, dinv, agg, E, 128);
        k_combine<<<N, 128, 0, stream>>>(h, agg, dinv, bc1, xb, 128, 78);
        // layer2: xb(128) -> h(192)
        gemm(xb, Wc2t, nullptr, h, N, 128, 192, 192, 192, 0, 0, 1, 0);
        zeroW(agg, (size_t)N * 192);
        k_scatter<<<E, 128, 0, stream>>>(h, ei, dinv, agg, E, 192);
        k_combine<<<N, 128, 0, stream>>>(h, agg, dinv, bc2, xa, 192, 156);
        // layer3: xa(192) -> h(320)
        gemm(xa, Wc3t, nullptr, h, N, 192, 320, 320, 320, 0, 0, 1, 0);
        zeroW(agg, (size_t)N * 320);
        k_scatter<<<E, 128, 0, stream>>>(h, ei, dinv, agg, E, 320);
        k_combine<<<N, 128, 0, stream>>>(h, agg, dinv, bc3, xb, 320, 312);
        // global max pool (1024 segments of 40 nodes)
        zeroW(gmax, (size_t)B * 320);
        k_pool<<<N, 128, 0, stream>>>(xb, gmax, 320, N / B);
        cvt(gmax, gmaxb, (size_t)B * 320, 0);
        // FC head
        gemm(gmaxb, Wg1t, bg1, t1, B, 320, 192, 192, 192, 156, 1, 1, 0);
        cvt(t1, t1b, (size_t)B * 192, 0);
        gemm(t1b, Wg2t, bg2, gout, B, 192, 128, 128, 128, 128, 0, 1, 0);
    };
    branch(x1, ei1, g1);
    branch(x2, ei2, g2);

    // ======== fusion ========
    k_concat<<<(B * 384 + 255) / 256, 256, 0, stream>>>(g1, g2, cvv, xcb, B);
    gemm(xcb, Wf1t, bf1_, f1, B, 384, 512, 512, 512, 512, 1, 1, 0);
    cvt(f1, f1b, (size_t)B * 512, 0);
    gemm(f1b, Wf2t, bf2_, f2, B, 512, 128, 128, 128, 128, 1, 1, 0);
    cvt(f2, f2b, (size_t)B * 128, 0);
    gemm(f2b, Wot, bo, out, B, 128, 64, /*ldc=*/2, /*nact=*/2, /*nbias=*/2, 0, 1, 0);
}
